// DeformableResidualBlock_30666066494194
// MI455X (gfx1250) — matrix operand order repaired and checked by reading, not compiled, not run on hardware
//
#include <hip/hip_runtime.h>
#include <hip/hip_bf16.h>

#define BATCH   4
#define CHN     256
#define HH      64
#define WW      64
#define HWX     (HH*WW)        // 4096
#define KKX     9
#define KDIM    (CHN*KKX)      // 2304
#define KTILES  (KDIM/32)      // 72
#define NTILES  (CHN/16)       // 16
#define GROUPS  32
#define CPG     (CHN/GROUPS)   // 8
#define GN_N    (CPG*HWX)      // 32768
#define EPSV    1e-5f
#define MPIX    32             // pixels per block (2 M-tiles)
#define MK2     (MPIX*KKX)     // 288 (pixel, kpoint) pairs per block

typedef __attribute__((ext_vector_type(16))) __bf16 v16bf;
typedef __attribute__((ext_vector_type(8)))  float  v8f;
typedef __attribute__((ext_vector_type(4)))  float  v4f;

__device__ __forceinline__ __bf16 f2bf(float f) {
  union { float f; unsigned u; } in; in.f = f;
  unsigned r = in.u + 0x7FFFu + ((in.u >> 16) & 1u);   // round-to-nearest-even
  unsigned short hh = (unsigned short)(r >> 16);
  __bf16 o; __builtin_memcpy(&o, &hh, 2); return o;
}

union FragBF { v16bf v; v4f q[2]; };

// ---------------------------------------------------------------------------
// Repack weights [O, C, 3, 3] (fp32) into bf16 WMMA B-fragments.
// Layout: [ntile][ktile][lane][16 bf16]; lane%16 = N column, K0 = 8*(lane/16);
// element i<8 -> K0+i, element i>=8 -> K0+16+(i-8) (matches A layout).
// GEMM K index kd = k9*256 + c  (A tile stored identically).
// ---------------------------------------------------------------------------
__global__ void pack_w_kernel(const float* __restrict__ w, __bf16* __restrict__ dst) {
  int idx = blockIdx.x * 256 + threadIdx.x;
  const int TOT = NTILES * KTILES * 32 * 16;
  if (idx >= TOT) return;
  int i    = idx & 15;
  int lane = (idx >> 4) & 31;
  int kt   = (idx >> 9) % KTILES;
  int nt   = idx / (512 * KTILES);
  int kloc = ((lane < 16) ? 0 : 8) + ((i < 8) ? i : i + 8);
  int kd   = kt * 32 + kloc;
  int k9   = kd >> 8;
  int c    = kd & 255;
  int o    = nt * 16 + (lane & 15);
  dst[idx] = f2bf(w[(o * CHN + c) * KKX + k9]);
}

// ---------------------------------------------------------------------------
// GroupNorm + ReLU. Input NCHW (flag 0) or NHWC (flag 1); output NHWC.
// ---------------------------------------------------------------------------
__global__ void gn_relu_kernel(const float* __restrict__ in, const float* __restrict__ g,
                               const float* __restrict__ bta, float* __restrict__ out,
                               int in_nhwc) {
  __shared__ float s0[256], s1[256];
  __shared__ float s_mean, s_rstd;
  int tid = threadIdx.x;
  int b   = blockIdx.x / GROUPS;
  int grp = blockIdx.x % GROUPS;
  float sum = 0.f, sq = 0.f;
  for (int idx = tid; idx < GN_N; idx += 256) {
    int cl  = idx >> 12;
    int pix = idx & (HWX - 1);
    int ch  = grp * CPG + cl;
    float v = in_nhwc ? in[(b * HWX + pix) * CHN + ch]
                      : in[(b * CHN + ch) * HWX + pix];
    sum += v; sq += v * v;
  }
  s0[tid] = sum; s1[tid] = sq; __syncthreads();
  for (int o = 128; o > 0; o >>= 1) {
    if (tid < o) { s0[tid] += s0[tid + o]; s1[tid] += s1[tid + o]; }
    __syncthreads();
  }
  if (tid == 0) {
    float m = s0[0] / (float)GN_N;
    float var = s1[0] / (float)GN_N - m * m;
    s_mean = m; s_rstd = rsqrtf(var + EPSV);
  }
  __syncthreads();
  float m = s_mean, rs = s_rstd;
  for (int idx = tid; idx < GN_N; idx += 256) {
    int cl  = idx >> 12;
    int pix = idx & (HWX - 1);
    int ch  = grp * CPG + cl;
    float v = in_nhwc ? in[(b * HWX + pix) * CHN + ch]
                      : in[(b * CHN + ch) * HWX + pix];
    float r = (v - m) * rs * g[ch] + bta[ch];
    out[(b * HWX + pix) * CHN + ch] = fmaxf(r, 0.f);
  }
}

// ---------------------------------------------------------------------------
// Fused depthwise 7x7 + 1x1 offset predictor -> absolute sample coords.
// One block (256 threads, thread=channel) per pixel. h is NHWC.
// ---------------------------------------------------------------------------
__global__ void offsets_kernel(const float* __restrict__ h, const float* __restrict__ dw,
                               const float* __restrict__ pw, const float* __restrict__ pwb,
                               float* __restrict__ py, float* __restrict__ px) {
  __shared__ float sF[CHN];
  int b   = blockIdx.x >> 12;
  int pix = blockIdx.x & (HWX - 1);
  int y = pix >> 6, x = pix & 63;
  int c = threadIdx.x;
  float acc = 0.f;
  for (int i = 0; i < 7; ++i) {
    int yy = y + i - 3;
    if ((unsigned)yy >= (unsigned)HH) continue;
    for (int j = 0; j < 7; ++j) {
      int xx = x + j - 3;
      if ((unsigned)xx >= (unsigned)WW) continue;
      acc += dw[c * 49 + i * 7 + j] * h[(b * HWX + yy * WW + xx) * CHN + c];
    }
  }
  sF[c] = acc;
  __syncthreads();
  if (c < 2 * KKX) {
    int j = c;
    float o = pwb[j];
    for (int cc = 0; cc < CHN; ++cc) o += pw[j * CHN + cc] * sF[cc];
    o *= 1.0f;                       // OFFSET_SCALE
    int kk = j >> 1;
    if ((j & 1) == 0) py[(b * KKX + kk) * HWX + pix] = (float)y + (float)(kk / 3 - 1) + o;
    else              px[(b * KKX + kk) * HWX + pix] = (float)x + (float)(kk % 3 - 1) + o;
  }
}

// ---------------------------------------------------------------------------
// Fused bilinear deformable gather + implicit GEMM via WMMA bf16.
// Block: 256 threads (8 waves). M = 32 pixels (2 M-tiles), N = 256 channels
// (each wave owns 2 N-tiles, both M-tiles -> 4 accumulators, 4 WMMAs/K-step).
// ---------------------------------------------------------------------------
__global__ void __launch_bounds__(256)
deform_wmma_kernel(const float* __restrict__ h, const float* __restrict__ py,
                   const float* __restrict__ px, const __bf16* __restrict__ wp,
                   const float* __restrict__ bias, const float* __restrict__ resid,
                   float* __restrict__ out, int add_resid_nchw) {
  extern __shared__ __align__(16) char smem[];
  __bf16* sA = (__bf16*)smem;                    // [MPIX][KDIM] bf16 A-tile
  __shared__ int   sIdx[4][MK2];
  __shared__ float sWt [4][MK2];

  int tid = threadIdx.x;
  int b   = blockIdx.x / (HWX / MPIX);
  int p0  = (blockIdx.x % (HWX / MPIX)) * MPIX;

  // Phase 0: bilinear tap setup for 32 pixels x 9 kernel points
  for (int t0 = tid; t0 < MK2; t0 += 256) {
    int m = t0 / KKX, k = t0 % KKX;
    int pix = p0 + m;
    float fy = py[(b * KKX + k) * HWX + pix];
    float fx = px[(b * KKX + k) * HWX + pix];
    float y0 = floorf(fy), x0 = floorf(fx);
    float wy1 = fy - y0, wx1 = fx - x0;
    float wy0 = 1.f - wy1, wx0 = 1.f - wx1;
#pragma unroll
    for (int t = 0; t < 4; ++t) {
      float yi = y0 + (float)(t >> 1);
      float xi = x0 + (float)(t & 1);
      bool valid = (yi >= 0.f) && (yi <= (float)(HH - 1)) &&
                   (xi >= 0.f) && (xi <= (float)(WW - 1));
      int yc = (int)fminf(fmaxf(yi, 0.f), (float)(HH - 1));
      int xc = (int)fminf(fmaxf(xi, 0.f), (float)(WW - 1));
      float wt = ((t >> 1) ? wy1 : wy0) * ((t & 1) ? wx1 : wx0);
      sIdx[t][t0] = (b * HWX + yc * WW + xc) * CHN;
      sWt [t][t0] = valid ? wt : 0.f;
    }
  }
  __syncthreads();

  // Phase 1: channel-parallel gather into LDS A-tile (kd = k*256 + c)
  {
    int c = tid;
    for (int mk = 0; mk < MK2; ++mk) {
      float a = sWt[0][mk] * h[sIdx[0][mk] + c]
              + sWt[1][mk] * h[sIdx[1][mk] + c]
              + sWt[2][mk] * h[sIdx[2][mk] + c]
              + sWt[3][mk] * h[sIdx[3][mk] + c];
      int m = mk / KKX, k = mk - m * KKX;
      sA[m * KDIM + k * CHN + c] = f2bf(a);
    }
  }
  __syncthreads();

  // Phase 2: WMMA GEMM. A lane layout: lanes 0-15 -> M=lane, K0=0;
  // lanes 16-31 -> M=lane-16, K0=8; chunks [K0..K0+7] and [K0+16..K0+23].
  int wave = tid >> 5, lane = tid & 31;
  int mrow = lane & 15;
  int k0   = (lane >> 4) * 8;
  int nt0 = wave, nt1 = wave + 8;
  v8f acc00 = {}; v8f acc01 = {};    // M-tile 0 x {nt0, nt1}
  v8f acc10 = {}; v8f acc11 = {};    // M-tile 1 x {nt0, nt1}

  const __bf16* sA0 = sA + mrow * KDIM + k0;
  const __bf16* sA1 = sA0 + 16 * KDIM;

  for (int kt = 0; kt < KTILES; ++kt) {
    FragBF a0, a1;
    const v4f* ap0 = (const v4f*)(sA0 + kt * 32);
    a0.q[0] = ap0[0];            // +0  elems (16 B)
    a0.q[1] = ap0[2];            // +16 elems (32 B)
    const v4f* ap1 = (const v4f*)(sA1 + kt * 32);
    a1.q[0] = ap1[0];
    a1.q[1] = ap1[2];

    FragBF b0, b1;
    const v4f* bp0 = (const v4f*)(wp + (size_t)((nt0 * KTILES + kt) * 32 + lane) * 16);
    b0.q[0] = bp0[0]; b0.q[1] = bp0[1];
    const v4f* bp1 = (const v4f*)(wp + (size_t)((nt1 * KTILES + kt) * 32 + lane) * 16);
    b1.q[0] = bp1[0]; b1.q[1] = bp1[1];

    if (kt + 1 < KTILES) {
      __builtin_prefetch(wp + (size_t)((nt0 * KTILES + kt + 1) * 32 + lane) * 16, 0, 3);
      __builtin_prefetch(wp + (size_t)((nt1 * KTILES + kt + 1) * 32 + lane) * 16, 0, 3);
    }

    acc00 = __builtin_amdgcn_wmma_f32_16x16x32_bf16(false, a0.v, false, b0.v,
                                                    (short)0, acc00, false, false);
    acc01 = __builtin_amdgcn_wmma_f32_16x16x32_bf16(false, a0.v, false, b1.v,
                                                    (short)0, acc01, false, false);
    acc10 = __builtin_amdgcn_wmma_f32_16x16x32_bf16(false, a1.v, false, b0.v,
                                                    (short)0, acc10, false, false);
    acc11 = __builtin_amdgcn_wmma_f32_16x16x32_bf16(false, a1.v, false, b1.v,
                                                    (short)0, acc11, false, false);
  }

  // Epilogue: C/D layout: VGPR v, lanes 0-15 -> M=v, lanes 16-31 -> M=v+8
  int n     = lane & 15;
  int mbase = (lane >> 4) * 8;
  float bia0 = bias[nt0 * 16 + n];
  float bia1 = bias[nt1 * 16 + n];
#pragma unroll
  for (int v = 0; v < 8; ++v) {
    int m    = v + mbase;
    int pix0 = p0 + m;          // M-tile 0
    int pix1 = p0 + 16 + m;     // M-tile 1
    int ch0  = nt0 * 16 + n, ch1 = nt1 * 16 + n;
    float r00 = acc00[v] + bia0;
    float r01 = acc01[v] + bia1;
    float r10 = acc10[v] + bia0;
    float r11 = acc11[v] + bia1;
    if (add_resid_nchw) {
      int i00 = (b * CHN + ch0) * HWX + pix0;
      int i01 = (b * CHN + ch1) * HWX + pix0;
      int i10 = (b * CHN + ch0) * HWX + pix1;
      int i11 = (b * CHN + ch1) * HWX + pix1;
      out[i00] = r00 + resid[i00];
      out[i01] = r01 + resid[i01];
      out[i10] = r10 + resid[i10];
      out[i11] = r11 + resid[i11];
    } else {
      out[(b * HWX + pix0) * CHN + ch0] = r00;
      out[(b * HWX + pix0) * CHN + ch1] = r01;
      out[(b * HWX + pix1) * CHN + ch0] = r10;
      out[(b * HWX + pix1) * CHN + ch1] = r11;
    }
  }
}

// ---------------------------------------------------------------------------
extern "C" void kernel_launch(void* const* d_in, const int* in_sizes, int n_in,
                              void* d_out, int out_size, void* d_ws, size_t ws_size,
                              hipStream_t stream) {
  (void)in_sizes; (void)n_in; (void)out_size; (void)ws_size;
  const float* x    = (const float*)d_in[0];
  const float* gn1g = (const float*)d_in[1];
  const float* gn1b = (const float*)d_in[2];
  const float* dw1  = (const float*)d_in[3];
  const float* pw1  = (const float*)d_in[4];
  const float* pwb1 = (const float*)d_in[5];
  const float* w1   = (const float*)d_in[6];
  const float* b1   = (const float*)d_in[7];
  const float* gn2g = (const float*)d_in[8];
  const float* gn2b = (const float*)d_in[9];
  const float* dw2  = (const float*)d_in[10];
  const float* pw2  = (const float*)d_in[11];
  const float* pwb2 = (const float*)d_in[12];
  const float* w2   = (const float*)d_in[13];
  const float* b2   = (const float*)d_in[14];
  float* out = (float*)d_out;

  char* ws = (char*)d_ws;
  size_t off = 0;
  float*  h   = (float*)(ws + off);  off += (size_t)BATCH * HWX * CHN * 4;
  float*  mid = (float*)(ws + off);  off += (size_t)BATCH * HWX * CHN * 4;
  float*  py  = (float*)(ws + off);  off += (size_t)BATCH * KKX * HWX * 4;
  float*  px  = (float*)(ws + off);  off += (size_t)BATCH * KKX * HWX * 4;
  __bf16* wp1 = (__bf16*)(ws + off); off += (size_t)NTILES * KTILES * 32 * 16 * 2;
  __bf16* wp2 = (__bf16*)(ws + off); off += (size_t)NTILES * KTILES * 32 * 16 * 2;

  const int packTot = NTILES * KTILES * 32 * 16;
  pack_w_kernel<<<(packTot + 255) / 256, 256, 0, stream>>>(w1, wp1);
  pack_w_kernel<<<(packTot + 255) / 256, 256, 0, stream>>>(w2, wp2);

  const int dynLds = MPIX * KDIM * 2;   // 147456 B

  // Pass 1
  gn_relu_kernel<<<BATCH * GROUPS, 256, 0, stream>>>(x, gn1g, gn1b, h, 0);
  offsets_kernel<<<BATCH * HWX, 256, 0, stream>>>(h, dw1, pw1, pwb1, py, px);
  deform_wmma_kernel<<<BATCH * HWX / MPIX, 256, dynLds, stream>>>(
      h, py, px, wp1, b1, x, mid, 0);

  // Pass 2 (+ residual, NCHW output)
  gn_relu_kernel<<<BATCH * GROUPS, 256, 0, stream>>>(mid, gn2g, gn2b, h, 1);
  offsets_kernel<<<BATCH * HWX, 256, 0, stream>>>(h, dw2, pw2, pwb2, py, px);
  deform_wmma_kernel<<<BATCH * HWX / MPIX, 256, dynLds, stream>>>(
      h, py, px, wp2, b2, x, out, 1);
}
